// EnhancedFDConv_12592844111991
// MI455X (gfx1250) — matrix operand order repaired and checked by reading, not compiled, not run on hardware
//
#include <hip/hip_runtime.h>
#include <math.h>

// ---------------------------------------------------------------------------
// MI455X / gfx1250 implementation. wave32, WMMA 16x16x32 f16 -> f32.
// ---------------------------------------------------------------------------

typedef float    v8f  __attribute__((ext_vector_type(8)));
typedef _Float16 v16h __attribute__((ext_vector_type(16)));
typedef _Float16 h8   __attribute__((ext_vector_type(8)));

#define B_   8
#define C_   64
#define O_   64
#define H_   256
#define W_   256
#define NPIX 65536

// ===========================================================================
// Generic implicit-GEMM conv with WMMA. All spatial dims compile-time so all
// index math lowers to shifts (no scalar division sequences).
//   - weights pre-packed f16 [COUT][K], K = ci*KH*KW + kh*KW + kw (flat OIHW)
//   - input may be a channel-concat of two tensors (in0: CSPLIT ch, in1: rest)
//   - per-batch weights (PERB) for the dynamic final conv
//   - ACT: 0 none, 1 relu, 2 sigmoid (bias always added)
// Block: 256 threads = 8 waves, tiled (8/NT) m-tiles x NT n-tiles.
// Each wave: one 16x16 output tile, K swept in KC-sized LDS-resident chunks.
// ===========================================================================
template<int CIN,int COUT,int KH,int KW,int PAD,int KC,int ACT,bool PERB,int CSPLIT>
__global__ __launch_bounds__(256,1) void conv_wmma_k(
    const float* __restrict__ in0, const float* __restrict__ in1,
    const _Float16* __restrict__ wpack, const float* __restrict__ bias,
    float* __restrict__ out)
{
  constexpr int K  = CIN*KH*KW;
  constexpr int NT = COUT/16;        // n-tiles per block
  constexpr int MT = 8/NT;           // m-tiles per block
  constexpr int PM = MT*16;          // pixels per block (contiguous along W)
  constexpr int BPR = W_/PM;         // blocks per image row
  constexpr int ROWBLK = H_*BPR;     // blocks per batch
  static_assert(K % KC == 0, "KC must divide K");

  extern __shared__ __align__(16) _Float16 smem[];   // [PM][KC] im2col (f16)

  const int b   = blockIdx.x / ROWBLK;               // constant divisor
  const int r0  = blockIdx.x - b*ROWBLK;
  const int oy  = r0 / BPR;
  const int ox0 = (r0 - oy*BPR) * PM;

  const int tid  = threadIdx.x;
  const int wv   = tid >> 5, lane = tid & 31;
  const int mt   = wv / NT, nt = wv - (wv/NT)*NT;
  const int l16  = lane & 15, hh = lane >> 4;
  const int n    = nt*16 + l16;        // output channel of this lane (B/C frag)
  const int mrow = mt*16 + l16;        // pixel row of this lane (A frag)

  const _Float16* wrow = wpack + (PERB ? (size_t)b*COUT*K : (size_t)0)
                               + (size_t)n*K;

  v8f acc = {0.f,0.f,0.f,0.f,0.f,0.f,0.f,0.f};

  for (int kc = 0; kc < K; kc += KC) {
    // warm next chunk's weight stream while this one computes
    if (kc + KC < K)
      __builtin_prefetch(wrow + kc + KC + 16*hh, 0, 3);

    // -------- cooperative im2col chunk into LDS: smem[m*KC + kk] ----------
    for (int e = tid; e < PM*KC; e += 256) {
      int m  = e / KC;
      int kk = e - m*KC;
      int k  = kc + kk;
      int ci = k / (KH*KW);
      int rr = k - ci*(KH*KW);
      int kh = rr / KW;
      int kw = rr - kh*KW;
      int iy = oy + kh - PAD;
      int ix = ox0 + m + kw - PAD;
      float v = 0.f;
      if (iy >= 0 && iy < H_ && ix >= 0 && ix < W_) {
        const float* src; int cc, nch;
        if (ci < CSPLIT) { src = in0; cc = ci;           nch = CSPLIT; }
        else             { src = in1; cc = ci - CSPLIT;  nch = CIN - CSPLIT; }
        v = src[(((size_t)b*nch + cc) << 16) + (iy << 8) + ix];
      }
      smem[m*KC + kk] = (_Float16)v;
    }
    __syncthreads();

    // -------- WMMA sweep over this chunk ----------------------------------
    #pragma unroll
    for (int ks = 0; ks < KC; ks += 32) {
      // A fragment (16x32 f16): lane(mrow,hh) holds K-runs [8h,8h+8) and
      // [16+8h,16+8h+8) -> two contiguous b128 LDS loads.
      h8 a0 = *(const h8*)&smem[mrow*KC + ks + 8*hh];
      h8 a1 = *(const h8*)&smem[mrow*KC + ks + 16 + 8*hh];
      v16h av = __builtin_shufflevector(a0, a1,
                 0,1,2,3,4,5,6,7,8,9,10,11,12,13,14,15);
      // B fragment (32x16 f16): lane(n,hh) holds K-runs [8h,8h+8) and
      // [16+8h,16+8h+8) -> two contiguous 16B loads from packed weights.
      h8 b0 = *(const h8*)(wrow + kc + ks + 8*hh);
      h8 b1 = *(const h8*)(wrow + kc + ks + 16 + 8*hh);
      v16h bv = __builtin_shufflevector(b0, b1,
                 0,1,2,3,4,5,6,7,8,9,10,11,12,13,14,15);
      acc = __builtin_amdgcn_wmma_f32_16x16x32_f16(
                false, av, false, bv, (short)0, acc, false, false);
    }
    __syncthreads();
  }

  // -------- epilogue: C frag lane(n,hh), VGPR r -> pixel mt*16 + r + 8h ---
  float bb = bias ? bias[n] : 0.f;
  #pragma unroll
  for (int r = 0; r < 8; r++) {
    int m = mt*16 + r + 8*hh;
    float v = acc[r] + bb;
    if (ACT == 1) v = v > 0.f ? v : 0.f;
    else if (ACT == 2) v = 1.f/(1.f + __expf(-v));
    out[(((size_t)b*COUT + n) << 16) + (oy << 8) + ox0 + m] = v;
  }
}

// ===========================================================================
// Elementwise / small kernels
// ===========================================================================
__global__ __launch_bounds__(256) void pack_f16_k(const float* __restrict__ s,
                                                  _Float16* __restrict__ d, int n) {
  int i = blockIdx.x*256 + threadIdx.x;
  if (i < n) d[i] = (_Float16)s[i];
}

__global__ void zero_k(float* p, int n) {
  int i = blockIdx.x*blockDim.x + threadIdx.x;
  if (i < n) p[i] = 0.f;
}

// enhanced = x*(1+anomaly) + 0.5*|sobel_dw(x)+edge_b|
__global__ __launch_bounds__(256) void enhance_k(
    const float* __restrict__ x, const float* __restrict__ anomaly,
    const float* __restrict__ edge_b, float* __restrict__ enhanced)
{
  size_t idx = (size_t)blockIdx.x*256 + threadIdx.x;   // over B*C*H*W
  int xx = idx & (W_-1);
  int yy = (idx >> 8) & (H_-1);
  int c  = (idx >> 16) & (C_-1);
  const float* img = x + (idx & ~(size_t)(NPIX-1));    // channel-plane base
  const float kx[9] = {-1,0,1,-2,0,2,-1,0,1};
  const float ky[9] = {-1,-2,-1,0,0,0,1,2,1};
  const float* kk = (c & 1) ? ky : kx;
  float acc = 0.f;
  #pragma unroll
  for (int t = 0; t < 9; t++) {
    int dy = t/3 - 1, dx = t%3 - 1;
    int iy = yy + dy, ix = xx + dx;
    if (iy >= 0 && iy < H_ && ix >= 0 && ix < W_) acc += kk[t]*img[iy*W_ + ix];
  }
  float edge = fabsf(acc + edge_b[c]);
  float xv = img[yy*W_ + xx];
  enhanced[idx] = xv*(1.f + anomaly[idx]) + 0.5f*edge;
}

// per-(b,c) spatial mean
__global__ __launch_bounds__(256) void gap_k(const float* __restrict__ in,
                                             float* __restrict__ g) {
  __shared__ float s[256];
  const float* p = in + (size_t)blockIdx.x*NPIX;
  float acc = 0.f;
  for (int i = threadIdx.x; i < NPIX; i += 256) acc += p[i];
  s[threadIdx.x] = acc; __syncthreads();
  for (int st = 128; st > 0; st >>= 1) {
    if (threadIdx.x < st) s[threadIdx.x] += s[threadIdx.x + st];
    __syncthreads();
  }
  if (threadIdx.x == 0) g[blockIdx.x] = s[0] * (1.f/NPIX);
}

// channel attention weights: cw = sigmoid(ca2(relu(ca1(g))))   (64 thr/blk)
__global__ void cw_k(const float* g1, const float* w1, const float* b1,
                     const float* w2, const float* b2, float* cw) {
  int b = blockIdx.x, t = threadIdx.x;
  __shared__ float hid[4];
  if (t < 4) {
    float s = b1[t];
    for (int c = 0; c < 64; c++) s += w1[t*64 + c]*g1[b*64 + c];
    hid[t] = s > 0.f ? s : 0.f;
  }
  __syncthreads();
  float s = b2[t];
  for (int j = 0; j < 4; j++) s += w2[t*4 + j]*hid[j];
  cw[b*64 + t] = 1.f/(1.f + __expf(-s));
}

// freq importance = softmax(fs2(relu(fs1(g)))/4)               (16 thr/blk)
__global__ void imp_k(const float* g2, const float* w1, const float* b1,
                      const float* w2, const float* b2, float* imp) {
  int b = blockIdx.x, t = threadIdx.x;
  __shared__ float hid[16], lg[8];
  float s = b1[t];
  for (int c = 0; c < 64; c++) s += w1[t*64 + c]*g2[b*64 + c];
  hid[t] = s > 0.f ? s : 0.f;
  __syncthreads();
  if (t < 8) {
    float s2 = b2[t];
    for (int j = 0; j < 16; j++) s2 += w2[t*16 + j]*hid[j];
    lg[t] = s2 * 0.25f;           // TEMP = 4
  }
  __syncthreads();
  if (t == 0) {
    float m = lg[0];
    for (int i = 1; i < 8; i++) m = fmaxf(m, lg[i]);
    float e[8], sum = 0.f;
    for (int i = 0; i < 8; i++) { e[i] = __expf(lg[i]-m); sum += e[i]; }
    for (int i = 0; i < 8; i++) imp[b*8 + i] = e[i]/sum;
  }
}

// pooled mean/max over channels of xe*cw
__global__ __launch_bounds__(256) void pooled_k(const float* __restrict__ xe,
                                                const float* __restrict__ cw,
                                                float* __restrict__ pooled) {
  size_t idx = (size_t)blockIdx.x*256 + threadIdx.x;   // over B*NPIX
  int b = (int)(idx >> 16);
  int pix = (int)(idx & (NPIX-1));
  const float* base = xe + (((size_t)b*64) << 16) + pix;
  float mn = 0.f, mx = -3.4e38f;
  for (int c = 0; c < 64; c++) {
    float v = base[(size_t)c << 16] * cw[b*64 + c];
    mn += v; mx = fmaxf(mx, v);
  }
  pooled[(((size_t)b*2    ) << 16) + pix] = mn * (1.f/64.f);
  pooled[(((size_t)b*2 + 1) << 16) + pix] = mx;
}

// spatial attention 7x7 conv on pooled, then xe *= cw * sigmoid(sa) in-place
__global__ __launch_bounds__(256) void sa_k(float* __restrict__ xe,
                                            const float* __restrict__ pooled,
                                            const float* __restrict__ cw,
                                            const float* __restrict__ sw,
                                            const float* __restrict__ sb) {
  size_t idx = (size_t)blockIdx.x*256 + threadIdx.x;   // over B*NPIX
  int b = (int)(idx >> 16);
  int pix = (int)(idx & (NPIX-1));
  int y = pix >> 8, x = pix & 255;
  float s = sb[0];
  for (int ch = 0; ch < 2; ch++) {
    const float* pb = pooled + (((size_t)b*2 + ch) << 16);
    for (int ky = 0; ky < 7; ky++) {
      int iy = y + ky - 3; if (iy < 0 || iy >= H_) continue;
      for (int kx = 0; kx < 7; kx++) {
        int ix = x + kx - 3; if (ix < 0 || ix >= W_) continue;
        s += sw[(ch*7 + ky)*7 + kx] * pb[iy*W_ + ix];
      }
    }
  }
  s = 1.f/(1.f + __expf(-s));
  float* base = xe + (((size_t)b*64) << 16) + pix;
  for (int c = 0; c < 64; c++) base[(size_t)c << 16] *= cw[b*64 + c]*s;
}

// pd2 1x1 (32->4) + per-pixel softmax + spatial-mean accumulation into pwacc
__global__ __launch_bounds__(256) void pd2_k(const float* __restrict__ t1p,
                                             const float* __restrict__ w,
                                             const float* __restrict__ bias,
                                             float* __restrict__ pwacc) {
  size_t idx = (size_t)blockIdx.x*256 + threadIdx.x;   // over B*NPIX
  int b = (int)(idx >> 16);
  int pix = (int)(idx & (NPIX-1));
  const float* base = t1p + (((size_t)b*32) << 16) + pix;
  float lg[4];
  #pragma unroll
  for (int p = 0; p < 4; p++) {
    float s = bias[p];
    for (int c = 0; c < 32; c++) s += w[p*32 + c]*base[(size_t)c << 16];
    lg[p] = s;
  }
  float m = fmaxf(fmaxf(lg[0],lg[1]), fmaxf(lg[2],lg[3]));
  float e[4], sum = 0.f;
  #pragma unroll
  for (int p = 0; p < 4; p++) { e[p] = __expf(lg[p]-m); sum += e[p]; }
  __shared__ float red[256];
  for (int p = 0; p < 4; p++) {
    red[threadIdx.x] = e[p]/sum; __syncthreads();
    for (int st = 128; st > 0; st >>= 1) {
      if (threadIdx.x < st) red[threadIdx.x] += red[threadIdx.x + st];
      __syncthreads();
    }
    if (threadIdx.x == 0) atomicAdd(&pwacc[b*4 + p], red[0]);
    __syncthreads();
  }
}

// dynamic per-batch kernels: wfin[b][o][k] = sum_p pw[b][p]/NPIX * pk[p][o][k]
__global__ __launch_bounds__(256) void dynw_k(const float* __restrict__ pwacc,
                                              const float* __restrict__ pk,
                                              _Float16* __restrict__ wfin) {
  int idx = blockIdx.x*256 + threadIdx.x;              // B*O*576
  if (idx >= B_*O_*576) return;
  int k = idx % 576;
  int o = (idx/576) % O_;
  int b = idx/(576*O_);
  float s = 0.f;
  #pragma unroll
  for (int p = 0; p < 4; p++)
    s += pwacc[b*4 + p]*(1.f/NPIX)*pk[((size_t)(p*O_ + o))*576 + k];
  wfin[idx] = (_Float16)s;
}

// ===========================================================================
// 256-pt Stockham FFT in LDS (128 threads = N/2 butterflies per stage).
// sgn = -1 forward, +1 inverse (unscaled). Data starts/ends in buffer 0.
// ===========================================================================
__device__ __forceinline__ int fft256_stages(float (*re)[256], float (*im)[256],
                                             int t, float sgn) {
  int cur = 0;
  #pragma unroll
  for (int s = 0; s < 8; s++) {
    int Ns = 1 << s;
    int q = t & (Ns - 1);
    int base = ((t >> s) << (s + 1)) + q;
    float ang = sgn * 3.14159265358979323846f * (float)q / (float)Ns;
    float sn, cs;
    __sincosf(ang, &sn, &cs);
    float ar = re[cur][t],      ai = im[cur][t];
    float br = re[cur][t+128],  bi = im[cur][t+128];
    float tr = cs*br - sn*bi,   ti = cs*bi + sn*br;
    int nx = cur ^ 1;
    re[nx][base]      = ar + tr; im[nx][base]      = ai + ti;
    re[nx][base + Ns] = ar - tr; im[nx][base + Ns] = ai - ti;
    cur = nx;
    __syncthreads();
  }
  return cur;
}

// forward row rFFT: real row (256) -> 129 complex bins
__global__ __launch_bounds__(128) void fft_row_fwd(const float* __restrict__ xe,
                                                   float2* __restrict__ spec) {
  __shared__ float re[2][256], im[2][256];
  size_t row = blockIdx.x;                    // (b*C + c)*H + y
  const float* src = xe + row*W_;
  int t = threadIdx.x;
  re[0][t] = src[t]; re[0][t+128] = src[t+128];
  im[0][t] = 0.f;    im[0][t+128] = 0.f;
  __syncthreads();
  int cur = fft256_stages(re, im, t, -1.f);
  float2* dst = spec + row*129;
  dst[t] = make_float2(re[cur][t], im[cur][t]);
  if (t == 0) dst[128] = make_float2(re[cur][128], im[cur][128]);
}

// column FFT + radial band mask + inverse column FFT (in place, 1/256 scale)
__global__ __launch_bounds__(128) void fft_col_filter(float2* __restrict__ spec,
                                                      const float* __restrict__ imp,
                                                      const float* __restrict__ thr) {
  __shared__ float re[2][256], im[2][256];
  int x  = blockIdx.x % 129;
  int bc = blockIdx.x / 129;
  int b  = bc >> 6;                           // / C_
  float2* col = spec + (size_t)bc*H_*129 + x;
  int t = threadIdx.x;
  float2 v0 = col[(size_t)t*129];
  float2 v1 = col[(size_t)(t+128)*129];
  re[0][t] = v0.x; im[0][t] = v0.y;
  re[0][t+128] = v1.x; im[0][t+128] = v1.y;
  __syncthreads();
  int cur = fft256_stages(re, im, t, -1.f);
  float fx = (float)x * (1.f/256.f);          // rfftfreq
  float t0=thr[0],t1=thr[1],t2=thr[2],t3=thr[3],t4=thr[4],t5=thr[5],t6=thr[6];
  #pragma unroll
  for (int u = 0; u < 2; u++) {
    int ky = t + 128*u;
    float fy = (ky < 128 ? (float)ky : (float)(ky - 256)) * (1.f/256.f);
    float f = sqrtf(fx*fx + fy*fy);
    int band = (f>t0)+(f>t1)+(f>t2)+(f>t3)+(f>t4)+(f>t5)+(f>t6);
    float w = imp[b*8 + band];
    re[cur][ky] *= w; im[cur][ky] *= w;
  }
  __syncthreads();
  cur = fft256_stages(re, im, t, 1.f);
  col[(size_t)t*129]       = make_float2(re[cur][t]*(1.f/256.f),
                                         im[cur][t]*(1.f/256.f));
  col[(size_t)(t+128)*129] = make_float2(re[cur][t+128]*(1.f/256.f),
                                         im[cur][t+128]*(1.f/256.f));
}

// inverse row FFT with Hermitian reconstruction -> real output (1/256 scale)
__global__ __launch_bounds__(128) void fft_row_inv(const float2* __restrict__ spec,
                                                   float* __restrict__ xf) {
  __shared__ float re[2][256], im[2][256];
  size_t row = blockIdx.x;
  const float2* src = spec + row*129;
  int t = threadIdx.x;
  float2 a = src[t];
  re[0][t] = a.x; im[0][t] = a.y;
  float2 c2;
  if (t == 0) c2 = src[128];
  else { c2 = src[128 - t]; c2.y = -c2.y; }   // conj symmetry for k=129..255
  re[0][t+128] = c2.x; im[0][t+128] = c2.y;
  __syncthreads();
  int cur = fft256_stages(re, im, t, 1.f);
  float* dst = xf + row*W_;
  dst[t]     = re[cur][t]     * (1.f/256.f);
  dst[t+128] = re[cur][t+128] * (1.f/256.f);
}

// ===========================================================================
// Host-side orchestration
// ===========================================================================
static inline size_t alignup(size_t v, size_t a) { return (v + a - 1) & ~(a - 1); }

extern "C" void kernel_launch(void* const* d_in, const int* in_sizes, int n_in,
                              void* d_out, int out_size, void* d_ws, size_t ws_size,
                              hipStream_t stream) {
  const float* x      = (const float*)d_in[0];
  const float* edge_b = (const float*)d_in[1];
  const float* an1_w  = (const float*)d_in[2];
  const float* an1_b  = (const float*)d_in[3];
  const float* an2_w  = (const float*)d_in[4];
  const float* an2_b  = (const float*)d_in[5];
  const float* fus_w  = (const float*)d_in[6];
  const float* fus_b  = (const float*)d_in[7];
  const float* ca1_w  = (const float*)d_in[8];
  const float* ca1_b  = (const float*)d_in[9];
  const float* ca2_w  = (const float*)d_in[10];
  const float* ca2_b  = (const float*)d_in[11];
  const float* sa_w   = (const float*)d_in[12];
  const float* sa_b   = (const float*)d_in[13];
  const float* fs1_w  = (const float*)d_in[14];
  const float* fs1_b  = (const float*)d_in[15];
  const float* fs2_w  = (const float*)d_in[16];
  const float* fs2_b  = (const float*)d_in[17];
  const float* thr    = (const float*)d_in[18];
  const float* pd1_w  = (const float*)d_in[19];
  const float* pd1_b  = (const float*)d_in[20];
  const float* pd2_w  = (const float*)d_in[21];
  const float* pd2_b  = (const float*)d_in[22];
  const float* pk     = (const float*)d_in[23];
  const float* bias   = (const float*)d_in[24];
  float* out = (float*)d_out;

  // ---- workspace layout (buffers reused across phases) --------------------
  char* base = (char*)d_ws;
  size_t off = 0;
  const size_t SZ_A = (size_t)B_*32*NPIX*4;          // 64 MB  (an1 out / pd1 out)
  const size_t SZ_B = (size_t)B_*64*NPIX*4;          // 128 MB (anomaly -> x_freq)
  const size_t SZ_C = (size_t)B_*64*H_*129*8;        // 129 MB (enhanced -> spectrum)
  const size_t SZ_D = SZ_B;                          // 128 MB (xe)
  float* bufA = (float*)(base + off); off = alignup(off + SZ_A, 256);
  float* bufB = (float*)(base + off); off = alignup(off + SZ_B, 256);
  float* bufC = (float*)(base + off); off = alignup(off + SZ_C, 256);
  float* bufD = (float*)(base + off); off = alignup(off + SZ_D, 256);
  float* pooled = (float*)(base + off); off = alignup(off + (size_t)B_*2*NPIX*4, 256);
  float* g1    = (float*)(base + off); off = alignup(off + 2048, 256);
  float* g2    = (float*)(base + off); off = alignup(off + 2048, 256);
  float* cw    = (float*)(base + off); off = alignup(off + 2048, 256);
  float* imp   = (float*)(base + off); off = alignup(off + 256, 256);
  float* pwacc = (float*)(base + off); off = alignup(off + 256, 256);
  _Float16* pan1 = (_Float16*)(base + off); off = alignup(off + 32*576*2, 256);
  _Float16* pan2 = (_Float16*)(base + off); off = alignup(off + 64*288*2, 256);
  _Float16* pfus = (_Float16*)(base + off); off = alignup(off + 64*128*2, 256);
  _Float16* ppd1 = (_Float16*)(base + off); off = alignup(off + 32*1600*2, 256);
  _Float16* wfin = (_Float16*)(base + off); off = alignup(off + (size_t)B_*64*576*2, 256);
  if (off > ws_size) return;   // insufficient scratch

  // ---- weight packing (f32 OIHW -> f16 [COUT][K]) -------------------------
  pack_f16_k<<<(32*576  + 255)/256, 256, 0, stream>>>(an1_w, pan1, 32*576);
  pack_f16_k<<<(64*288  + 255)/256, 256, 0, stream>>>(an2_w, pan2, 64*288);
  pack_f16_k<<<(64*128  + 255)/256, 256, 0, stream>>>(fus_w, pfus, 64*128);
  pack_f16_k<<<(32*1600 + 255)/256, 256, 0, stream>>>(pd1_w, ppd1, 32*1600);
  zero_k<<<1, 32, 0, stream>>>(pwacc, 32);

  // ---- defect enhancer ----------------------------------------------------
  // t1 = relu(an1(x))                      [B,32,H,W] -> bufA
  conv_wmma_k<64,32,3,3,1,288,1,false,64><<<B_*H_*4, 256, 64*288*2, stream>>>(
      x, x, pan1, an1_b, bufA);
  // anomaly = sigmoid(an2(t1))             [B,64,H,W] -> bufB
  conv_wmma_k<32,64,3,3,1,288,2,false,32><<<B_*H_*8, 256, 32*288*2, stream>>>(
      bufA, bufA, pan2, an2_b, bufB);
  // enhanced = x*(1+anomaly)+0.5*|sobel(x)+b|          -> bufC
  enhance_k<<<(B_*C_*NPIX)/256, 256, 0, stream>>>(x, bufB, edge_b, bufC);
  // xe = fus([x ; enhanced])  (1x1, K=128)  [B,64,H,W] -> bufD
  conv_wmma_k<128,64,1,1,0,128,0,false,64><<<B_*H_*8, 256, 32*128*2, stream>>>(
      x, bufC, pfus, fus_b, bufD);

  // ---- texture pattern path (uses x only; bufA free now) ------------------
  conv_wmma_k<64,32,5,5,2,320,1,false,64><<<B_*H_*4, 256, 64*320*2, stream>>>(
      x, x, ppd1, pd1_b, bufA);
  pd2_k<<<(B_*NPIX)/256, 256, 0, stream>>>(bufA, pd2_w, pd2_b, pwacc);

  // ---- channel attention --------------------------------------------------
  gap_k<<<B_*C_, 256, 0, stream>>>(bufD, g1);
  cw_k<<<B_, 64, 0, stream>>>(g1, ca1_w, ca1_b, ca2_w, ca2_b, cw);

  // ---- spatial attention (xe updated in place) ----------------------------
  pooled_k<<<(B_*NPIX)/256, 256, 0, stream>>>(bufD, cw, pooled);
  sa_k<<<(B_*NPIX)/256, 256, 0, stream>>>(bufD, pooled, cw, sa_w, sa_b);

  // ---- frequency importance ----------------------------------------------
  gap_k<<<B_*C_, 256, 0, stream>>>(bufD, g2);
  imp_k<<<B_, 16, 0, stream>>>(g2, fs1_w, fs1_b, fs2_w, fs2_b, imp);

  // ---- adaptive frequency filtering (rfft2 * mask -> irfft2) --------------
  fft_row_fwd<<<B_*C_*H_, 128, 0, stream>>>(bufD, (float2*)bufC);
  fft_col_filter<<<B_*C_*129, 128, 0, stream>>>((float2*)bufC, imp, thr);
  fft_row_inv<<<B_*C_*H_, 128, 0, stream>>>((const float2*)bufC, bufB); // x_freq

  // ---- dynamic per-batch kernels + final conv -----------------------------
  dynw_k<<<(B_*O_*576 + 255)/256, 256, 0, stream>>>(pwacc, pk, wfin);
  conv_wmma_k<64,64,3,3,1,288,0,true,64><<<B_*H_*8, 256, 32*288*2, stream>>>(
      bufB, bufB, wfin, bias, out);
}
